// AttnBlock_54778012893275
// MI455X (gfx1250) — matrix operand order repaired and checked by reading, not compiled, not run on hardware
//
#include <hip/hip_runtime.h>
#include <stdint.h>

#define B_ 8
#define C_ 512
#define N_ 2048
#define G_ 32

typedef __attribute__((ext_vector_type(16))) __bf16 v16bf;
typedef __attribute__((ext_vector_type(8)))  float  v8f;

union Frag {
  v16bf v;
  uint4 q[2];
  __bf16 h[16];
};

static __device__ __forceinline__ __bf16 f2bf(float f) {
  unsigned u = __builtin_bit_cast(unsigned, f);
  unsigned r = (u + 0x7FFFu + ((u >> 16) & 1u)) >> 16;
  unsigned short s = (unsigned short)r;
  return __builtin_bit_cast(__bf16, s);
}

// A fragment (16x32 bf16): h[0..7] = K[ksel..ksel+7], h[8..15] = K[16+ksel..16+ksel+7]
static __device__ __forceinline__ void loadA(Frag& f, const __bf16* p) {
  f.q[0] = *(const uint4*)p;
  f.q[1] = *(const uint4*)(p + 16);
}
// B fragment (32x16 bf16): h[0..7] = K[kbB..kbB+7], h[8..15] = K[16+kbB..16+kbB+7] at fixed column
static __device__ __forceinline__ void loadB(Frag& f, const __bf16* p) {
  f.q[0] = *(const uint4*)p;
  f.q[1] = *(const uint4*)(p + 16);
}

#define WMMA_BF16(A, Bb, Cc) \
  __builtin_amdgcn_wmma_f32_16x16x32_bf16(false, (A), false, (Bb), (short)0, (Cc), false, false)

// ---------------- weight fp32 -> bf16 ----------------
__global__ void wconv_kernel(const float* wq, const float* wk, const float* wv,
                             const float* wp, __bf16* wbf) {
  int idx = blockIdx.x * 256 + threadIdx.x;           // 4 * 512 * 512 total
  int which = idx >> 18;
  int off = idx & 0x3FFFF;
  const float* s = which == 0 ? wq : which == 1 ? wk : which == 2 ? wv : wp;
  wbf[idx] = f2bf(s[off]);
}

// ---------------- group-norm statistics ----------------
__global__ void gn_stats_kernel(const float* x, float* gnstat) {
  int bg = blockIdx.x;                                // b*32+g ; slab is contiguous
  const float* p = x + (size_t)bg * (16 * N_);
  float s = 0.f, ss = 0.f;
  for (int i = threadIdx.x; i < 16 * N_; i += 256) {
    float v = p[i];
    s += v; ss += v * v;
  }
  __shared__ float sh1[256], sh2[256];
  sh1[threadIdx.x] = s; sh2[threadIdx.x] = ss;
  __syncthreads();
  for (int st = 128; st > 0; st >>= 1) {
    if (threadIdx.x < st) {
      sh1[threadIdx.x] += sh1[threadIdx.x + st];
      sh2[threadIdx.x] += sh2[threadIdx.x + st];
    }
    __syncthreads();
  }
  if (threadIdx.x == 0) {
    const float inv = 1.0f / (16.0f * N_);
    float mean = sh1[0] * inv;
    float var  = sh2[0] * inv - mean * mean;
    gnstat[2 * bg]     = mean;
    gnstat[2 * bg + 1] = rsqrtf(var + 1e-6f);
  }
}

// ---------------- group-norm apply + transpose to hT[B][N][C] bf16 ----------------
__global__ void gn_apply_kernel(const float* x, const float* gsc, const float* gbi,
                                const float* gnstat, __bf16* hT) {
  int n0 = blockIdx.x * 64, c0 = blockIdx.y * 16, b = blockIdx.z;
  int bg = b * G_ + (c0 >> 4);
  float mean = gnstat[2 * bg], ist = gnstat[2 * bg + 1];
  __shared__ float t[16][65];
  int col = threadIdx.x & 63, r0 = threadIdx.x >> 6;
#pragma unroll
  for (int k = 0; k < 4; ++k) {
    int row = r0 + 4 * k;
    int c = c0 + row;
    float v = x[((size_t)b * C_ + c) * N_ + n0 + col];
    t[row][col] = (v - mean) * ist * gsc[c] + gbi[c];
  }
  __syncthreads();
  int c = threadIdx.x & 15, nn0 = threadIdx.x >> 4;
#pragma unroll
  for (int k = 0; k < 4; ++k) {
    int n = nn0 + 16 * k;
    hT[((size_t)b * N_ + n0 + n) * C_ + c0 + c] = f2bf(t[c][n]);
  }
}

// ---------------- q/k/v conv1x1 GEMM (single-wave blocks) ----------------
__global__ void __launch_bounds__(32)
qkv_gemm_kernel(const __bf16* hT, const __bf16* wbf,
                const float* bq, const float* bk, const float* bv,
                __bf16* qT, __bf16* kT, __bf16* vN) {
  int l = threadIdx.x;
  int n0 = blockIdx.x * 64, m0 = blockIdx.y * 16;
  int which = blockIdx.z % 3, b = blockIdx.z / 3;
  const __bf16* w = wbf + (size_t)which * (C_ * C_);
  const float* bias = which == 0 ? bq : which == 1 ? bk : bv;
  int lm = l & 15, half = l >> 4;
  int ksel = half * 8, kbB = half * 8, off = half * 8;
  v8f acc[4] = {};
  const __bf16* arow = w + (size_t)(m0 + lm) * C_;
#pragma unroll 4
  for (int kb = 0; kb < C_; kb += 32) {
    __builtin_prefetch((const void*)(arow + kb + 128), 0, 1);  // global_prefetch
    Frag a; loadA(a, arow + kb + ksel);
#pragma unroll
    for (int nt = 0; nt < 4; ++nt) {
      int n = n0 + nt * 16 + lm;
      Frag bb; loadB(bb, hT + ((size_t)b * N_ + n) * C_ + kb + kbB);
      acc[nt] = WMMA_BF16(a.v, bb.v, acc[nt]);
    }
  }
  float br[8];
#pragma unroll
  for (int r = 0; r < 8; ++r) br[r] = bias[m0 + off + r];
  if (which < 2) {
    __bf16* out = which == 0 ? qT : kT;   // position-major [B][N][C]
#pragma unroll
    for (int nt = 0; nt < 4; ++nt) {
      int n = n0 + nt * 16 + lm;
      Frag o;
#pragma unroll
      for (int r = 0; r < 8; ++r) o.h[r] = f2bf(acc[nt][r] + br[r]);
      *(uint4*)(out + ((size_t)b * N_ + n) * C_ + m0 + off) = o.q[0];
    }
  } else {                                // v natural [B][C][N]
#pragma unroll
    for (int nt = 0; nt < 4; ++nt) {
      int n = n0 + nt * 16 + lm;
#pragma unroll
      for (int r = 0; r < 8; ++r)
        vN[((size_t)b * C_ + m0 + off + r) * N_ + n] = f2bf(acc[nt][r] + br[r]);
    }
  }
}

// ---------------- softmax pass 1: row max + sum-exp (single-wave blocks) ----------------
__global__ void __launch_bounds__(32)
attn_stats_kernel(const __bf16* qT, const __bf16* kT, float* sm, float* sil) {
  int l = threadIdx.x;
  int i0 = blockIdx.x * 16, b = blockIdx.y;
  int lm = l & 15, half = l >> 4;
  int ksel = half * 8, kbB = half * 8, off = half * 8;
  const float scale = 0.044194173824159216f;  // 512^-0.5
  const __bf16* qrow = qT + ((size_t)b * N_ + i0 + lm) * C_;
  float mr[8], lr[8];
#pragma unroll
  for (int r = 0; r < 8; ++r) { mr[r] = -3.0e38f; lr[r] = 0.f; }
#pragma unroll 1
  for (int j0 = 0; j0 < N_; j0 += 16) {
    v8f s = {};
    const __bf16* krow = kT + ((size_t)b * N_ + j0 + lm) * C_;
#pragma unroll
    for (int kb = 0; kb < C_; kb += 32) {
      Frag a; loadA(a, qrow + kb + ksel);
      Frag bb; loadB(bb, krow + kb + kbB);
      s = WMMA_BF16(a.v, bb.v, s);
    }
#pragma unroll
    for (int r = 0; r < 8; ++r) {
      float sv = s[r] * scale;
      float mn = fmaxf(mr[r], sv);
      lr[r] = lr[r] * __expf(mr[r] - mn) + __expf(sv - mn);
      mr[r] = mn;
    }
  }
  // merge across the 16-lane half holding the same rows
#pragma unroll
  for (int mask = 8; mask >= 1; mask >>= 1) {
#pragma unroll
    for (int r = 0; r < 8; ++r) {
      float mo = __shfl_xor(mr[r], mask, 32);
      float lo = __shfl_xor(lr[r], mask, 32);
      float mn = fmaxf(mr[r], mo);
      lr[r] = lr[r] * __expf(mr[r] - mn) + lo * __expf(mo - mn);
      mr[r] = mn;
    }
  }
  if (lm == 0) {
#pragma unroll
    for (int r = 0; r < 8; ++r) {
      sm[(size_t)b * N_ + i0 + off + r]  = mr[r];
      sil[(size_t)b * N_ + i0 + off + r] = 1.0f / lr[r];
    }
  }
}

// ---------------- attention output: probs (LDS) x V, flash-style (8 waves) ----------------
__global__ void __launch_bounds__(256)
attn_out_kernel(const __bf16* qT, const __bf16* kT, const __bf16* vN,
                const float* sm, const float* sil, __bf16* attnT) {
  int tid = threadIdx.x, w = tid >> 5, l = tid & 31;
  int lm = l & 15, half = l >> 4;
  int ksel = half * 8, kbB = half * 8, off = half * 8;
  int i0 = blockIdx.x * 16, b = blockIdx.y;
  const float scale = 0.044194173824159216f;
  __shared__ __attribute__((aligned(16))) __bf16 probs[8 * 16 * 40]; // [strip][row i][32 j + pad]
  const __bf16* qrow = qT + ((size_t)b * N_ + i0 + lm) * C_;
  float mrow[8];
#pragma unroll
  for (int r = 0; r < 8; ++r) mrow[r] = sm[(size_t)b * N_ + i0 + off + r];
  v8f acc[4] = {};
  int cw0 = w * 64;          // this wave's 64-channel output range
#pragma unroll 1
  for (int js = 0; js < N_; js += 256) {
    // each wave produces probabilities for its 32-wide j strip
#pragma unroll
    for (int t = 0; t < 2; ++t) {
      int jb = js + w * 32 + t * 16;
      v8f s = {};
      const __bf16* krow = kT + ((size_t)b * N_ + jb + lm) * C_;
#pragma unroll
      for (int kb = 0; kb < C_; kb += 32) {
        Frag a; loadA(a, qrow + kb + ksel);
        Frag bb; loadB(bb, krow + kb + kbB);
        s = WMMA_BF16(a.v, bb.v, s);
      }
      int jloc = t * 16 + lm;
#pragma unroll
      for (int r = 0; r < 8; ++r)
        probs[w * 640 + (off + r) * 40 + jloc] = f2bf(__expf(s[r] * scale - mrow[r]));
    }
    __syncthreads();
    // every wave consumes all 8 strips as A fragments against its V columns
#pragma unroll 1
    for (int jblk = 0; jblk < 8; ++jblk) {
      Frag a;
      const __bf16* pr = &probs[jblk * 640 + lm * 40 + ksel];
      a.q[0] = *(const uint4*)pr;
      a.q[1] = *(const uint4*)(pr + 16);
#pragma unroll
      for (int ct = 0; ct < 4; ++ct) {
        int c = cw0 + ct * 16 + lm;
        Frag bb; loadB(bb, vN + ((size_t)b * C_ + c) * N_ + js + jblk * 32 + kbB);
        acc[ct] = WMMA_BF16(a.v, bb.v, acc[ct]);
      }
    }
    __syncthreads();
  }
  float il[8];
#pragma unroll
  for (int r = 0; r < 8; ++r) il[r] = sil[(size_t)b * N_ + i0 + off + r];
#pragma unroll
  for (int ct = 0; ct < 4; ++ct)
#pragma unroll
    for (int r = 0; r < 8; ++r)
      attnT[((size_t)b * N_ + i0 + off + r) * C_ + cw0 + ct * 16 + lm] =
          f2bf(acc[ct][r] * il[r]);
}

// ---------------- projection conv1x1 + bias + residual ----------------
__global__ void __launch_bounds__(32)
proj_kernel(const __bf16* attnT, const __bf16* wbf, const float* bp,
            const float* x, float* out) {
  int l = threadIdx.x;
  int n0 = blockIdx.x * 64, m0 = blockIdx.y * 16, b = blockIdx.z;
  int lm = l & 15, half = l >> 4;
  int ksel = half * 8, kbB = half * 8, off = half * 8;
  const __bf16* w = wbf + (size_t)3 * C_ * C_;
  v8f acc[4] = {};
  const __bf16* arow = w + (size_t)(m0 + lm) * C_;
#pragma unroll 4
  for (int kb = 0; kb < C_; kb += 32) {
    Frag a; loadA(a, arow + kb + ksel);
#pragma unroll
    for (int nt = 0; nt < 4; ++nt) {
      int n = n0 + nt * 16 + lm;
      Frag bb; loadB(bb, attnT + ((size_t)b * N_ + n) * C_ + kb + kbB);
      acc[nt] = WMMA_BF16(a.v, bb.v, acc[nt]);
    }
  }
  float br[8];
#pragma unroll
  for (int r = 0; r < 8; ++r) br[r] = bp[m0 + off + r];
#pragma unroll
  for (int nt = 0; nt < 4; ++nt) {
    int n = n0 + nt * 16 + lm;
#pragma unroll
    for (int r = 0; r < 8; ++r) {
      size_t idx = ((size_t)b * C_ + m0 + off + r) * N_ + n;
      out[idx] = x[idx] + acc[nt][r] + br[r];
    }
  }
}

extern "C" void kernel_launch(void* const* d_in, const int* in_sizes, int n_in,
                              void* d_out, int out_size, void* d_ws, size_t ws_size,
                              hipStream_t stream) {
  (void)in_sizes; (void)n_in; (void)out_size; (void)ws_size;
  const float* x   = (const float*)d_in[0];
  const float* gsc = (const float*)d_in[1];
  const float* gbi = (const float*)d_in[2];
  const float* wq  = (const float*)d_in[3];
  const float* bq  = (const float*)d_in[4];
  const float* wk  = (const float*)d_in[5];
  const float* bk  = (const float*)d_in[6];
  const float* wv  = (const float*)d_in[7];
  const float* bv  = (const float*)d_in[8];
  const float* wp  = (const float*)d_in[9];
  const float* bp  = (const float*)d_in[10];
  float* out = (float*)d_out;

  char* ws = (char*)d_ws;
  const size_t SZ = (size_t)B_ * N_ * C_ * 2;     // one bf16 [B,*,*] tensor = 16 MiB
  __bf16* hT    = (__bf16*)(ws);
  __bf16* qT    = (__bf16*)(ws + SZ);
  __bf16* kT    = (__bf16*)(ws + 2 * SZ);
  __bf16* vN    = (__bf16*)(ws + 3 * SZ);
  __bf16* attnT = (__bf16*)(ws + 4 * SZ);
  __bf16* wbf   = (__bf16*)(ws + 5 * SZ);
  char* tail    = ws + 5 * SZ + (size_t)4 * C_ * C_ * 2;
  float* gnstat = (float*)tail;                    // 256 * 2 floats
  float* sm     = (float*)(tail + 4096);           // [B,N]
  float* sil    = sm + (size_t)B_ * N_;            // [B,N]

  wconv_kernel<<<4096, 256, 0, stream>>>(wq, wk, wv, wp, wbf);
  gn_stats_kernel<<<B_ * G_, 256, 0, stream>>>(x, gnstat);
  gn_apply_kernel<<<dim3(N_ / 64, C_ / 16, B_), 256, 0, stream>>>(x, gsc, gbi, gnstat, hT);
  qkv_gemm_kernel<<<dim3(N_ / 64, C_ / 16, 3 * B_), 32, 0, stream>>>(hT, wbf, bq, bk, bv, qT, kT, vN);
  attn_stats_kernel<<<dim3(N_ / 16, B_), 32, 0, stream>>>(qT, kT, sm, sil);
  attn_out_kernel<<<dim3(N_ / 16, B_), 256, 0, stream>>>(qT, kT, vN, sm, sil, attnT);
  proj_kernel<<<dim3(N_ / 64, C_ / 16, B_), 32, 0, stream>>>(attnT, wbf, bp, x, out);
}
